// DualSelfAttention_12987981103217
// MI455X (gfx1250) — compile-verified, matrix operand order checked by reading, not run on hardware
//
#include <hip/hip_runtime.h>
#include <hip/hip_bf16.h>

// ---------------------------------------------------------------------------
// Dual self-attention for MI455X (gfx1250, wave32, WMMA f16 -> f32 accum).
// Flash-style two-pass position attention (never materializes the 256 MB
// attention matrix); channel attention via WMMA gram + small VALU GEMM.
// ---------------------------------------------------------------------------

typedef __attribute__((ext_vector_type(16))) _Float16 v16h;
typedef __attribute__((ext_vector_type(8)))  _Float16 v8h;
typedef __attribute__((ext_vector_type(8)))  float    v8f;
typedef _Float16 half_t;

#define B_  4
#define C_  64
#define H_  64
#define W_  64
#define N_  4096   // H_*W_

// Load a 16-lane-half K-fragment: lane owns one row (A) / column (B) whose
// 16 f16 elements live at k = k0 + {hi8..hi8+7} and k0 + {16+hi8..16+hi8+7}
// (CDNA5 16-bit 16x32 operand layout). Two aligned 16B chunks per lane.
static __device__ __forceinline__ v16h load_frag(const half_t* rowbase, int k0, int hi8) {
  v8h a = *(const v8h*)(rowbase + k0 + hi8);
  v8h b = *(const v8h*)(rowbase + k0 + 16 + hi8);
  v16h r;
#pragma unroll
  for (int e = 0; e < 8; ++e) { r[e] = a[e]; r[e + 8] = b[e]; }
  return r;
}

static __device__ __forceinline__ v8f wmma16(v16h a, v16h b, v8f c) {
  return __builtin_amdgcn_wmma_f32_16x16x32_f16(false, a, false, b, (short)0, c,
                                                false, false);
}

// ---------------------------------------------------------------------------
// x (f32) -> xb (f16), same [b][c][n] layout
// ---------------------------------------------------------------------------
__global__ __launch_bounds__(256) void pack_x_kernel(const float* __restrict__ x,
                                                     half_t* __restrict__ xb, int n) {
  int i = blockIdx.x * 256 + threadIdx.x;
  if (i < n) xb[i] = (half_t)x[i];
}

// ---------------------------------------------------------------------------
// 3x3 SAME convs for q,k,v.  One block per (batch,row): 3 input rows staged
// in LDS (48 KB), thread t handles out-channel t&63 and 16 pixels.
// Outputs: qT,kT as f16 [b][n][c] (WMMA-friendly), v as f16 [b][c][n].
// ---------------------------------------------------------------------------
__global__ __launch_bounds__(256) void conv_qkv_kernel(
    const float* __restrict__ x,
    const float* __restrict__ wq, const float* __restrict__ bq,
    const float* __restrict__ wk, const float* __restrict__ bk,
    const float* __restrict__ wv, const float* __restrict__ bv,
    half_t* __restrict__ qT, half_t* __restrict__ kT, half_t* __restrict__ vf) {
  __shared__ float xs[3][C_][W_];
  int b = blockIdx.x >> 6;
  int y = blockIdx.x & 63;
  int tid = threadIdx.x;
  for (int e = tid; e < 3 * C_ * W_; e += 256) {
    int ky = e / (C_ * W_);
    int rem = e - ky * (C_ * W_);
    int ci = rem >> 6, xx = rem & 63;
    int yy = y + ky - 1;
    float val = 0.f;
    if (yy >= 0 && yy < H_) val = x[(((size_t)b * C_ + ci) * H_ + yy) * W_ + xx];
    xs[ky][ci][xx] = val;
  }
  __syncthreads();

  int co = tid & 63;
  int xg = tid >> 6;  // 4 groups of 16 pixels
  float aq[16], ak[16], av[16];
#pragma unroll
  for (int p = 0; p < 16; ++p) { aq[p] = 0.f; ak[p] = 0.f; av[p] = 0.f; }

  for (int ci = 0; ci < C_; ++ci) {
    float wqr[9], wkr[9], wvr[9];
    const float* wq9 = wq + ((size_t)co * C_ + ci) * 9;
    const float* wk9 = wk + ((size_t)co * C_ + ci) * 9;
    const float* wv9 = wv + ((size_t)co * C_ + ci) * 9;
#pragma unroll
    for (int t = 0; t < 9; ++t) { wqr[t] = wq9[t]; wkr[t] = wk9[t]; wvr[t] = wv9[t]; }
#pragma unroll
    for (int p = 0; p < 16; ++p) {
      int xi = xg * 16 + p;
#pragma unroll
      for (int ky = 0; ky < 3; ++ky) {
#pragma unroll
        for (int kx = 0; kx < 3; ++kx) {
          int xx = xi + kx - 1;
          int xc = xx < 0 ? 0 : (xx > 63 ? 63 : xx);
          float val = (xx >= 0 && xx < W_) ? xs[ky][ci][xc] : 0.f;
          int t = ky * 3 + kx;
          aq[p] = fmaf(wqr[t], val, aq[p]);
          ak[p] = fmaf(wkr[t], val, ak[p]);
          av[p] = fmaf(wvr[t], val, av[p]);
        }
      }
    }
  }
  float bqv = bq[co], bkv = bk[co], bvv = bv[co];
#pragma unroll
  for (int p = 0; p < 16; ++p) {
    int xi = xg * 16 + p;
    int n = y * W_ + xi;
    qT[((size_t)b * N_ + n) * C_ + co] = (half_t)(aq[p] + bqv);
    kT[((size_t)b * N_ + n) * C_ + co] = (half_t)(ak[p] + bkv);
    vf[((size_t)(b * C_ + co)) * N_ + n] = (half_t)(av[p] + bvv);
  }
}

// ---------------------------------------------------------------------------
// Pass 1: per-row softmax statistics m[i], 1/Z[i] of S = Q^T K.
// Block = 64 rows (4 waves x 16 rows). Each wave streams all 4096 columns,
// computing 16x16 S tiles with WMMA, online (m,Z) per lane, then a
// shfl_xor butterfly merges the 16 lanes that share each row.
// ---------------------------------------------------------------------------
__global__ __launch_bounds__(128) void pos_stats_kernel(
    const half_t* __restrict__ qT, const half_t* __restrict__ kT,
    float* __restrict__ mOut, float* __restrict__ zInv) {
  int b = blockIdx.x >> 6;
  int ib = blockIdx.x & 63;
  int wave = threadIdx.x >> 5;
  int lane = threadIdx.x & 31;
  int col = lane & 15;
  int hi8 = (lane >> 4) << 3;  // 0 or 8
  int i0 = ib * 64 + wave * 16;

  const half_t* qrow = qT + ((size_t)b * N_ + i0 + col) * C_;
  v16h aq0 = load_frag(qrow, 0, hi8);
  v16h aq1 = load_frag(qrow, 32, hi8);

  float m[8], Z[8];
#pragma unroll
  for (int r = 0; r < 8; ++r) { m[r] = -3.0e38f; Z[r] = 0.f; }

  for (int jt = 0; jt < N_ / 16; ++jt) {
    int j0 = jt * 16;
    const half_t* krow = kT + ((size_t)b * N_ + j0 + col) * C_;
    v8f s = {};
    s = wmma16(aq0, load_frag(krow, 0, hi8), s);
    s = wmma16(aq1, load_frag(krow, 32, hi8), s);
#pragma unroll
    for (int r = 0; r < 8; ++r) {
      float sv = s[r];
      float mn = fmaxf(m[r], sv);
      Z[r] = Z[r] * __expf(m[r] - mn) + __expf(sv - mn);
      m[r] = mn;
    }
  }
#pragma unroll
  for (int mask = 1; mask <= 8; mask <<= 1) {
#pragma unroll
    for (int r = 0; r < 8; ++r) {
      float mo = __shfl_xor(m[r], mask);
      float zo = __shfl_xor(Z[r], mask);
      float mn = fmaxf(m[r], mo);
      Z[r] = Z[r] * __expf(m[r] - mn) + zo * __expf(mo - mn);
      m[r] = mn;
    }
  }
  if (col == 0) {
#pragma unroll
    for (int r = 0; r < 8; ++r) {
      int i = i0 + hi8 + r;
      mOut[(size_t)b * N_ + i] = m[r];
      zInv[(size_t)b * N_ + i] = 1.0f / Z[r];
    }
  }
}

// ---------------------------------------------------------------------------
// Pass 2: pos[c,j] = sum_i v[c,i] * exp(s[i,j]-m[i]) / Z[i].
// Block owns a 64-column j-block (wave -> 16 cols); loops i in steps of 32:
// recompute 2 S tiles via WMMA, build the P B-fragment entirely in registers
// (D-tile layout == B-fragment element set), then 4 WMMAs accumulate V x P.
// m/Zinv staged in 32 KB LDS, shared by all 4 waves.
// ---------------------------------------------------------------------------
__global__ __launch_bounds__(128) void pos_attn_kernel(
    const half_t* __restrict__ qT, const half_t* __restrict__ kT,
    const half_t* __restrict__ vf,
    const float* __restrict__ mRow, const float* __restrict__ zInv,
    float* __restrict__ pos) {
  __shared__ float mS[N_];
  __shared__ float zS[N_];
  int b = blockIdx.x >> 6;
  int jb = blockIdx.x & 63;
  int tid = threadIdx.x;
  for (int e = tid; e < N_; e += 128) {
    mS[e] = mRow[(size_t)b * N_ + e];
    zS[e] = zInv[(size_t)b * N_ + e];
  }
  __syncthreads();

  int wave = tid >> 5, lane = tid & 31;
  int col = lane & 15;
  int hi8 = (lane >> 4) << 3;
  int j0 = jb * 64 + wave * 16;

  const half_t* krow = kT + ((size_t)b * N_ + j0 + col) * C_;
  v16h bk0 = load_frag(krow, 0, hi8);
  v16h bk1 = load_frag(krow, 32, hi8);

  v8f oacc[4];
#pragma unroll
  for (int ct = 0; ct < 4; ++ct) oacc[ct] = {};

  for (int it = 0; it < N_ / 32; ++it) {
    int i0 = it * 32;
    const half_t* q0 = qT + ((size_t)b * N_ + i0 + col) * C_;
    const half_t* q1 = qT + ((size_t)b * N_ + i0 + 16 + col) * C_;
    v8f s0 = {}, s1 = {};
    s0 = wmma16(load_frag(q0, 0, hi8), bk0, s0);
    s0 = wmma16(load_frag(q0, 32, hi8), bk1, s0);
    s1 = wmma16(load_frag(q1, 0, hi8), bk0, s1);
    s1 = wmma16(load_frag(q1, 32, hi8), bk1, s1);

    v16h pf;
#pragma unroll
    for (int r = 0; r < 8; ++r) {
      int ia = i0 + hi8 + r;
      int ib2 = i0 + 16 + hi8 + r;
      pf[r]     = (half_t)(__expf(s0[r] - mS[ia]) * zS[ia]);
      pf[r + 8] = (half_t)(__expf(s1[r] - mS[ib2]) * zS[ib2]);
    }
#pragma unroll
    for (int ct = 0; ct < 4; ++ct) {
      const half_t* vrow = vf + ((size_t)(b * C_ + ct * 16 + col)) * N_;
      oacc[ct] = wmma16(load_frag(vrow, i0, hi8), pf, oacc[ct]);
    }
  }
#pragma unroll
  for (int ct = 0; ct < 4; ++ct) {
#pragma unroll
    for (int r = 0; r < 8; ++r) {
      int c = ct * 16 + hi8 + r;
      pos[((size_t)b * C_ + c) * N_ + j0 + col] = oacc[ct][r];
    }
  }
}

// ---------------------------------------------------------------------------
// Channel gram: G[b][c][d] = sum_n x[c,n] x[d,n]  (WMMA, K = 4096)
// ---------------------------------------------------------------------------
__global__ __launch_bounds__(128) void gram_kernel(const half_t* __restrict__ xb,
                                                   float* __restrict__ G) {
  int b = blockIdx.x >> 2;
  int ct = blockIdx.x & 3;
  int wave = threadIdx.x >> 5, lane = threadIdx.x & 31;
  int col = lane & 15;
  int hi8 = (lane >> 4) << 3;
  int c0 = ct * 16, d0 = wave * 16;
  const half_t* arow = xb + ((size_t)(b * C_ + c0 + col)) * N_;
  const half_t* brow = xb + ((size_t)(b * C_ + d0 + col)) * N_;
  v8f acc = {};
  for (int nt = 0; nt < N_ / 32; ++nt) {
    int n0 = nt * 32;
    acc = wmma16(load_frag(arow, n0, hi8), load_frag(brow, n0, hi8), acc);
  }
#pragma unroll
  for (int r = 0; r < 8; ++r) {
    int c = c0 + hi8 + r, d = d0 + col;
    G[((size_t)b * C_ + c) * C_ + d] = acc[r];
  }
}

__global__ __launch_bounds__(64) void chan_softmax_kernel(const float* __restrict__ G,
                                                          float* __restrict__ A) {
  int b = blockIdx.x;
  int c = threadIdx.x;
  const float* g = G + ((size_t)b * C_ + c) * C_;
  float mx = -3.0e38f;
  for (int d = 0; d < C_; ++d) mx = fmaxf(mx, g[d]);
  float s = 0.f;
  for (int d = 0; d < C_; ++d) s += __expf(g[d] - mx);
  float inv = 1.0f / s;
  float* a = A + ((size_t)b * C_ + c) * C_;
  for (int d = 0; d < C_; ++d) a[d] = __expf(g[d] - mx) * inv;
}

// ---------------------------------------------------------------------------
// Combine: out = pos_gamma*pos + chan_gamma*(Attn @ x) + 2*x
// Block = (batch, 64-pixel tile); attn (16 KB) and x tile (16 KB) in LDS.
// ---------------------------------------------------------------------------
__global__ __launch_bounds__(256) void combine_kernel(
    const float* __restrict__ x, const float* __restrict__ pos,
    const float* __restrict__ A,
    const float* __restrict__ pg, const float* __restrict__ cg,
    float* __restrict__ out) {
  __shared__ float aS[C_ * C_];
  __shared__ float xS[C_][64];
  int b = blockIdx.x >> 6;
  int n0 = (blockIdx.x & 63) * 64;
  int tid = threadIdx.x;
  for (int e = tid; e < C_ * C_; e += 256) aS[e] = A[(size_t)b * C_ * C_ + e];
  for (int e = tid; e < C_ * 64; e += 256) {
    int d = e >> 6, nl = e & 63;
    xS[d][nl] = x[((size_t)b * C_ + d) * N_ + n0 + nl];
  }
  __syncthreads();
  float pgv = pg[0], cgv = cg[0];
  int c = tid >> 2;
  int nl0 = (tid & 3) * 16;
  float acc[16];
#pragma unroll
  for (int k = 0; k < 16; ++k) acc[k] = 0.f;
  for (int d = 0; d < C_; ++d) {
    float a = aS[c * C_ + d];
#pragma unroll
    for (int k = 0; k < 16; ++k) acc[k] = fmaf(a, xS[d][nl0 + k], acc[k]);
  }
#pragma unroll
  for (int k = 0; k < 16; ++k) {
    int n = n0 + nl0 + k;
    float xv = xS[c][nl0 + k];
    size_t idx = ((size_t)b * C_ + c) * N_ + n;
    out[idx] = pgv * pos[idx] + cgv * acc[k] + 2.0f * xv;
  }
}

// ---------------------------------------------------------------------------
extern "C" void kernel_launch(void* const* d_in, const int* in_sizes, int n_in,
                              void* d_out, int out_size, void* d_ws, size_t ws_size,
                              hipStream_t stream) {
  (void)in_sizes; (void)n_in; (void)out_size; (void)ws_size;
  const float* x  = (const float*)d_in[0];
  const float* wq = (const float*)d_in[1];
  const float* bq = (const float*)d_in[2];
  const float* wk = (const float*)d_in[3];
  const float* bk = (const float*)d_in[4];
  const float* wv = (const float*)d_in[5];
  const float* bv = (const float*)d_in[6];
  const float* pg = (const float*)d_in[7];
  const float* cg = (const float*)d_in[8];
  float* out = (float*)d_out;

  char* ws = (char*)d_ws;
  size_t off = 0;
  const size_t elems = (size_t)B_ * C_ * N_;         // 1,048,576
  half_t* qT   = (half_t*)(ws + off); off += elems * sizeof(half_t);   // [b][n][c]
  half_t* kT   = (half_t*)(ws + off); off += elems * sizeof(half_t);   // [b][n][c]
  half_t* vf   = (half_t*)(ws + off); off += elems * sizeof(half_t);   // [b][c][n]
  half_t* xb   = (half_t*)(ws + off); off += elems * sizeof(half_t);   // [b][c][n]
  float*  mRow = (float*)(ws + off);  off += (size_t)B_ * N_ * sizeof(float);
  float*  zInv = (float*)(ws + off);  off += (size_t)B_ * N_ * sizeof(float);
  float*  pos  = (float*)(ws + off);  off += elems * sizeof(float);
  float*  G    = (float*)(ws + off);  off += (size_t)B_ * C_ * C_ * sizeof(float);
  float*  Attn = (float*)(ws + off);  off += (size_t)B_ * C_ * C_ * sizeof(float);

  pack_x_kernel<<<(int)((elems + 255) / 256), 256, 0, stream>>>(x, xb, (int)elems);
  conv_qkv_kernel<<<B_ * H_, 256, 0, stream>>>(x, wq, bq, wk, bk, wv, bv, qT, kT, vf);
  pos_stats_kernel<<<B_ * (N_ / 64), 128, 0, stream>>>(qT, kT, mRow, zInv);
  gram_kernel<<<B_ * 4, 128, 0, stream>>>(xb, G);
  chan_softmax_kernel<<<B_, 64, 0, stream>>>(G, Attn);
  pos_attn_kernel<<<B_ * (N_ / 64), 128, 0, stream>>>(qT, kT, vf, mRow, zInv, pos);
  combine_kernel<<<B_ * (N_ / 64), 256, 0, stream>>>(x, pos, Attn, pg, cg, out);
}
